// SPROF_GO_76776835383565
// MI455X (gfx1250) — matrix operand order repaired and checked by reading, not compiled, not run on hardware
//
#include <hip/hip_runtime.h>

typedef __attribute__((ext_vector_type(16))) _Float16 v16h;
typedef __attribute__((ext_vector_type(8)))  _Float16 v8h;
typedef __attribute__((ext_vector_type(8)))  float    v8f;

#define EPSLN 1e-6f
#define NEGSL 0.01f

// ---------------------------------------------------------------------------
// Pack f32 weight W[K x Nreal] (row-major) into WMMA B-fragment layout:
// tile (kt,nt) -> 512 halves; lane l holds N = nt*16 + (l&15),
// K = kt*32 + (l>=16 ? 16 : 0) + e  for e = 0..15 (contiguous per lane).
// ---------------------------------------------------------------------------
__global__ void pack_b_kernel(const float* __restrict__ W, _Float16* __restrict__ Bp,
                              int K, int Nreal, int NT) {
    const int kt = blockIdx.x, nt = blockIdx.y, lane = threadIdx.x;
    const int n = nt * 16 + (lane & 15);
    const int kbase = kt * 32 + ((lane >= 16) ? 16 : 0);
    _Float16* dst = Bp + ((size_t)(kt * NT + nt)) * 512 + lane * 16;
#pragma unroll
    for (int e = 0; e < 16; ++e) {
        int k = kbase + e;
        float v = (n < Nreal && k < K) ? W[(size_t)k * Nreal + n] : 0.0f;
        dst[e] = (_Float16)v;
    }
}

// ---------------------------------------------------------------------------
// Generic LayerNorm over width W (f32 in -> f16 out). One block per row.
// ---------------------------------------------------------------------------
__global__ void ln_kernel(const float* __restrict__ in, const float* __restrict__ g,
                          const float* __restrict__ b, _Float16* __restrict__ outH, int W) {
    const int row = blockIdx.x;
    const float* x = in + (size_t)row * W;
    __shared__ float ssum[256], ssq[256];
    float s = 0.f, q = 0.f;
    for (int i = threadIdx.x; i < W; i += 256) { float v = x[i]; s += v; q += v * v; }
    ssum[threadIdx.x] = s; ssq[threadIdx.x] = q;
    __syncthreads();
    for (int st = 128; st > 0; st >>= 1) {
        if (threadIdx.x < st) {
            ssum[threadIdx.x] += ssum[threadIdx.x + st];
            ssq[threadIdx.x]  += ssq[threadIdx.x + st];
        }
        __syncthreads();
    }
    float mean = ssum[0] / (float)W;
    float var  = ssq[0] / (float)W - mean * mean;
    float inv  = rsqrtf(var + EPSLN);
    for (int i = threadIdx.x; i < W; i += 256)
        outH[(size_t)row * W + i] = (_Float16)((x[i] - mean) * inv * g[i] + b[i]);
}

// ---------------------------------------------------------------------------
// Fused GEMM(+bias+leaky+LN) via v_wmma_f32_16x16x32_f16.
// One block = one 16-row tile of A[rows x K] against all N columns.
// waves = N/16/TPW, each wave owns TPW 16-col tiles. Row stats via ds_add_f32.
// ---------------------------------------------------------------------------
template <int TPW>
__global__ void gemm_ln_kernel(const _Float16* __restrict__ A, const _Float16* __restrict__ Bp,
                               const float* __restrict__ bias, const float* __restrict__ gamma,
                               const float* __restrict__ beta,
                               _Float16* __restrict__ outH, float* __restrict__ outF,
                               int K, int N) {
    const int lane  = threadIdx.x & 31;
    const int wave  = threadIdx.x >> 5;
    const int NT    = N >> 4;
    const int laneN = lane & 15;
    const int off8  = (lane >> 4) << 3;
    const size_t rowBase = (size_t)blockIdx.x * 16;

    __shared__ float s_sum[16], s_sq[16];
    if (threadIdx.x < 16) { s_sum[threadIdx.x] = 0.f; s_sq[threadIdx.x] = 0.f; }
    __syncthreads();

    v8f acc[TPW] = {};
    const _Float16* arow = A + (rowBase + laneN) * K + ((lane >> 4) << 4);
    const int KT = K >> 5;
    for (int kt = 0; kt < KT; ++kt) {
        v8h a0 = *(const v8h*)(arow + kt * 32);
        v8h a1 = *(const v8h*)(arow + kt * 32 + 8);
        v16h av;
#pragma unroll
        for (int i = 0; i < 8; ++i) { av[i] = a0[i]; av[i + 8] = a1[i]; }
#pragma unroll
        for (int t = 0; t < TPW; ++t) {
            const int nt = wave * TPW + t;
            v16h bv = *(const v16h*)(Bp + ((size_t)(kt * NT + nt)) * 512 + lane * 16);
            acc[t] = __builtin_amdgcn_wmma_f32_16x16x32_f16(
                false, av, false, bv, (short)0, acc[t], false, false);
        }
    }

    float vals[TPW][8];
#pragma unroll
    for (int t = 0; t < TPW; ++t) {
        const int col = (wave * TPW + t) * 16 + laneN;
        const float bcol = bias[col];
#pragma unroll
        for (int r = 0; r < 8; ++r) {
            float v = acc[t][r] + bcol;
            v = (v >= 0.f) ? v : NEGSL * v;
            vals[t][r] = v;
            const int m = r + off8;
            atomicAdd(&s_sum[m], v);
            atomicAdd(&s_sq[m], v * v);
        }
    }
    __syncthreads();

    const float invN = 1.0f / (float)N;
#pragma unroll
    for (int t = 0; t < TPW; ++t) {
        const int col = (wave * TPW + t) * 16 + laneN;
        const float g = gamma[col], bb = beta[col];
#pragma unroll
        for (int r = 0; r < 8; ++r) {
            const int m = r + off8;
            float mean = s_sum[m] * invN;
            float var  = s_sq[m] * invN - mean * mean;
            float y = (vals[t][r] - mean) * rsqrtf(var + EPSLN) * g + bb;
            size_t idx = (rowBase + m) * N + col;
            outH[idx] = (_Float16)y;
            if (outF) outF[idx] = y;
        }
    }
}

// ---------------------------------------------------------------------------
// Single-wave 16x16 WMMA tile GEMM (for the M=8-padded output block).
// actMode: 0 = leaky, 1 = sigmoid. Column-guarded for ragged N (4766).
// ---------------------------------------------------------------------------
__global__ void gemm_tile_kernel(const _Float16* __restrict__ A, const _Float16* __restrict__ Bp,
                                 const float* __restrict__ bias, float* __restrict__ out,
                                 int K, int NT, int Nreal, int ldOut, int rowsOut, int actMode) {
    const int lane  = threadIdx.x & 31;
    const int laneN = lane & 15;
    const int off8  = (lane >> 4) << 3;
    const int nt    = blockIdx.x;

    v8f acc = {};
    const _Float16* arow = A + (size_t)laneN * K + ((lane >> 4) << 4);
    const int KT = K >> 5;
    for (int kt = 0; kt < KT; ++kt) {
        v8h a0 = *(const v8h*)(arow + kt * 32);
        v8h a1 = *(const v8h*)(arow + kt * 32 + 8);
        v16h av;
#pragma unroll
        for (int i = 0; i < 8; ++i) { av[i] = a0[i]; av[i + 8] = a1[i]; }
        v16h bv = *(const v16h*)(Bp + ((size_t)(kt * NT + nt)) * 512 + lane * 16);
        acc = __builtin_amdgcn_wmma_f32_16x16x32_f16(
            false, av, false, bv, (short)0, acc, false, false);
    }

    const int col = nt * 16 + laneN;
    if (col < Nreal) {
        const float bcol = bias[col];
#pragma unroll
        for (int r = 0; r < 8; ++r) {
            const int m = r + off8;
            if (m < rowsOut) {
                float v = acc[r] + bcol;
                v = (actMode == 0) ? ((v >= 0.f) ? v : NEGSL * v)
                                   : 1.0f / (1.0f + __expf(-v));
                out[(size_t)m * ldOut + col] = v;
            }
        }
    }
}

// ---------------------------------------------------------------------------
// a2 = aln @ a_w2 + a_b2  (64 -> 8 heads, per row; tiny, VALU)
// ---------------------------------------------------------------------------
__global__ void logits_kernel(const _Float16* __restrict__ aln, const float* __restrict__ w2,
                              const float* __restrict__ b2, float* __restrict__ logits) {
    const int row = blockIdx.x * 256 + threadIdx.x;   // 0..8191
    const _Float16* a = aln + (size_t)row * 64;
    float acc[8];
#pragma unroll
    for (int h = 0; h < 8; ++h) acc[h] = b2[h];
    for (int k = 0; k < 64; ++k) {
        float av = (float)a[k];
#pragma unroll
        for (int h = 0; h < 8; ++h) acc[h] += av * w2[k * 8 + h];
    }
#pragma unroll
    for (int h = 0; h < 8; ++h) logits[(size_t)row * 8 + h] = acc[h];
}

// ---------------------------------------------------------------------------
// Masked softmax over L=1024 for one (b, head) per block.
// ---------------------------------------------------------------------------
__global__ void softmax_kernel(const float* __restrict__ logits, const int* __restrict__ mask,
                               float* __restrict__ asoft) {
    const int b = blockIdx.x >> 3, h = blockIdx.x & 7;
    __shared__ float red[256];
    float vals[4];
#pragma unroll
    for (int j = 0; j < 4; ++j) {
        int idx = b * 1024 + threadIdx.x + j * 256;
        float v = logits[(size_t)idx * 8 + h];
        if (mask[idx] == 0) v = -1e9f;
        vals[j] = v;
    }
    float mx = fmaxf(fmaxf(vals[0], vals[1]), fmaxf(vals[2], vals[3]));
    red[threadIdx.x] = mx; __syncthreads();
    for (int st = 128; st > 0; st >>= 1) {
        if (threadIdx.x < st) red[threadIdx.x] = fmaxf(red[threadIdx.x], red[threadIdx.x + st]);
        __syncthreads();
    }
    mx = red[0]; __syncthreads();
    float s = 0.f;
#pragma unroll
    for (int j = 0; j < 4; ++j) { vals[j] = __expf(vals[j] - mx); s += vals[j]; }
    red[threadIdx.x] = s; __syncthreads();
    for (int st = 128; st > 0; st >>= 1) {
        if (threadIdx.x < st) red[threadIdx.x] += red[threadIdx.x + st];
        __syncthreads();
    }
    const float inv = 1.0f / red[0];
#pragma unroll
    for (int j = 0; j < 4; ++j) {
        int idx = b * 1024 + threadIdx.x + j * 256;
        asoft[(size_t)idx * 8 + h] = vals[j] * inv;
    }
}

// ---------------------------------------------------------------------------
// pooled[b, h*256+d] = sum_l asoft[b,l,h] * x2f[b,l,d]; block per (b,h).
// ---------------------------------------------------------------------------
__global__ void pool_kernel(const float* __restrict__ asoft, const float* __restrict__ x2f,
                            float* __restrict__ pooled) {
    const int b = blockIdx.x >> 3, h = blockIdx.x & 7, d = threadIdx.x;
    const float* ab = asoft + (size_t)b * 1024 * 8 + h;
    const float* xb = x2f + (size_t)b * 1024 * 256 + d;
    float acc = 0.f;
    for (int l = 0; l < 1024; ++l) acc += ab[l * 8] * xb[l * 256];
    pooled[(size_t)b * 2048 + h * 256 + d] = acc;
}

// pooled f32 [8x2048] -> f16 [16x2048], rows 8..15 zeroed (WMMA M padding)
__global__ void cvtpad_kernel(const float* __restrict__ pooled, _Float16* __restrict__ A16) {
    const int i = blockIdx.x * 256 + threadIdx.x;   // 0..32767
    const int r = i >> 11, c = i & 2047;
    A16[i] = (r < 8) ? (_Float16)pooled[r * 2048 + c] : (_Float16)0.0f;
}

__global__ void zfill_kernel(_Float16* __restrict__ p, int n) {
    const int i = blockIdx.x * 256 + threadIdx.x;
    if (i < n) p[i] = (_Float16)0.0f;
}

// ---------------------------------------------------------------------------
// out[b,j] = max_k p[b,k] * CM[j,k].  One block per j; CM streamed once.
// ---------------------------------------------------------------------------
__global__ void maxprop_kernel(const float* __restrict__ p, const float* __restrict__ CM,
                               float* __restrict__ out) {
    const int j = blockIdx.x;
    const float* cmrow = CM + (size_t)j * 4766;
    float m[8];
#pragma unroll
    for (int b = 0; b < 8; ++b) m[b] = 0.f;
    for (int k = threadIdx.x; k < 4766; k += 256) {
        float c = cmrow[k];
        if (c != 0.0f) {
#pragma unroll
            for (int b = 0; b < 8; ++b) m[b] = fmaxf(m[b], p[(size_t)b * 4766 + k] * c);
        }
    }
    __shared__ float red[8][256];
#pragma unroll
    for (int b = 0; b < 8; ++b) red[b][threadIdx.x] = m[b];
    __syncthreads();
    for (int st = 128; st > 0; st >>= 1) {
        if (threadIdx.x < st) {
#pragma unroll
            for (int b = 0; b < 8; ++b)
                red[b][threadIdx.x] = fmaxf(red[b][threadIdx.x], red[b][threadIdx.x + st]);
        }
        __syncthreads();
    }
    if (threadIdx.x < 8) out[(size_t)threadIdx.x * 4766 + j] = red[threadIdx.x][0];
}

// ---------------------------------------------------------------------------
extern "C" void kernel_launch(void* const* d_in, const int* in_sizes, int n_in,
                              void* d_out, int out_size, void* d_ws, size_t ws_size,
                              hipStream_t stream) {
    (void)in_sizes; (void)n_in; (void)out_size; (void)ws_size;
    const float* h_V   = (const float*)d_in[0];
    const int*   mask  = (const int*)d_in[1];
    const float* ln0_g = (const float*)d_in[2];
    const float* ln0_b = (const float*)d_in[3];
    const float* w_in  = (const float*)d_in[4];
    const float* b_in  = (const float*)d_in[5];
    const float* ln1_g = (const float*)d_in[6];
    const float* ln1_b = (const float*)d_in[7];
    const float* w_h   = (const float*)d_in[8];
    const float* b_h   = (const float*)d_in[9];
    const float* ln2_g = (const float*)d_in[10];
    const float* ln2_b = (const float*)d_in[11];
    const float* a_w1  = (const float*)d_in[12];
    const float* a_b1  = (const float*)d_in[13];
    const float* a_lng = (const float*)d_in[14];
    const float* a_lnb = (const float*)d_in[15];
    const float* a_w2  = (const float*)d_in[16];
    const float* a_b2  = (const float*)d_in[17];
    const float* o_w1  = (const float*)d_in[18];
    const float* o_b1  = (const float*)d_in[19];
    const float* o_lng = (const float*)d_in[20];
    const float* o_lnb = (const float*)d_in[21];
    const float* o_w2  = (const float*)d_in[22];
    const float* o_b2  = (const float*)d_in[23];
    const float* CM    = (const float*)d_in[24];
    float* out = (float*)d_out;

    size_t off = 0;
    auto alloc = [&](size_t bytes) -> char* {
        char* p = (char*)d_ws + off;
        off += (bytes + 511) & ~(size_t)511;
        return p;
    };
    _Float16* x0     = (_Float16*)alloc(8192ull * 1024 * 2);
    _Float16* x1     = (_Float16*)alloc(8192ull * 256 * 2);
    _Float16* x2h    = (_Float16*)alloc(8192ull * 256 * 2);
    float*    x2f    = (float*)   alloc(8192ull * 256 * 4);
    _Float16* aln    = (_Float16*)alloc(8192ull * 64 * 2);
    float*    logits = (float*)   alloc(8192ull * 8 * 4);
    float*    asoft  = (float*)   alloc(8192ull * 8 * 4);
    float*    pooled = (float*)   alloc(8ull * 2048 * 4);
    _Float16* A16    = (_Float16*)alloc(16ull * 2048 * 2);
    float*    h1     = (float*)   alloc(16ull * 2048 * 4);
    _Float16* hln16  = (_Float16*)alloc(16ull * 2048 * 2);
    float*    pbuf   = (float*)   alloc(8ull * 4766 * 4);
    _Float16* pw_in  = (_Float16*)alloc(1024ull * 256 * 2);
    _Float16* pw_h   = (_Float16*)alloc(256ull * 256 * 2);
    _Float16* pa_w1  = (_Float16*)alloc(256ull * 64 * 2);
    _Float16* po_w1  = (_Float16*)alloc(2048ull * 2048 * 2);
    _Float16* po_w2  = (_Float16*)alloc(2048ull * 4768 * 2);

    // ---- weight packing into WMMA B-fragment layout (once per launch) ----
    pack_b_kernel<<<dim3(32, 16),  32, 0, stream>>>(w_in, pw_in, 1024, 256, 16);
    pack_b_kernel<<<dim3(8, 16),   32, 0, stream>>>(w_h,  pw_h,  256,  256, 16);
    pack_b_kernel<<<dim3(8, 4),    32, 0, stream>>>(a_w1, pa_w1, 256,  64,  4);
    pack_b_kernel<<<dim3(64, 128), 32, 0, stream>>>(o_w1, po_w1, 2048, 2048, 128);
    pack_b_kernel<<<dim3(64, 298), 32, 0, stream>>>(o_w2, po_w2, 2048, 4766, 298);

    // ---- trunk: LN0 -> [GEMM+leaky+LN] x2 -> attention MLP ----
    ln_kernel<<<8192, 256, 0, stream>>>(h_V, ln0_g, ln0_b, x0, 1024);
    gemm_ln_kernel<2><<<512, 256, 0, stream>>>(x0, pw_in, b_in, ln1_g, ln1_b,
                                               x1, nullptr, 1024, 256);
    gemm_ln_kernel<2><<<512, 256, 0, stream>>>(x1, pw_h, b_h, ln2_g, ln2_b,
                                               x2h, x2f, 256, 256);
    gemm_ln_kernel<1><<<512, 128, 0, stream>>>(x2h, pa_w1, a_b1, a_lng, a_lnb,
                                               aln, nullptr, 256, 64);
    logits_kernel<<<32, 256, 0, stream>>>(aln, a_w2, a_b2, logits);
    softmax_kernel<<<64, 256, 0, stream>>>(logits, mask, asoft);
    pool_kernel<<<64, 256, 0, stream>>>(asoft, x2f, pooled);

    // ---- output block (M=8 padded to 16) ----
    cvtpad_kernel<<<128, 256, 0, stream>>>(pooled, A16);
    gemm_tile_kernel<<<128, 32, 0, stream>>>(A16, po_w1, o_b1, h1,
                                             2048, 128, 2048, 2048, 16, 0);
    zfill_kernel<<<128, 256, 0, stream>>>(hln16, 16 * 2048);
    ln_kernel<<<8, 256, 0, stream>>>(h1, o_lng, o_lnb, hln16, 2048);
    gemm_tile_kernel<<<298, 32, 0, stream>>>(hln16, po_w2, o_b2, pbuf,
                                             2048, 298, 4766, 4766, 8, 1);

    // ---- GO-hierarchy max-product propagation ----
    maxprop_kernel<<<4766, 256, 0, stream>>>(pbuf, CM, out);
}
